// GRUEncDec_58102317580842
// MI455X (gfx1250) — matrix operand order repaired and checked by reading, not compiled, not run on hardware
//
#include <hip/hip_runtime.h>
#include <hip/hip_bf16.h>

#define B_TOTAL   131072
#define T_SEQ     51
#define HID       64
#define NOUT      20
#define DIM       4

typedef __attribute__((ext_vector_type(16))) __bf16 v16bf;
typedef __attribute__((ext_vector_type(8)))  __bf16 v8bf;
typedef __attribute__((ext_vector_type(4)))  __bf16 v4bf;
typedef __attribute__((ext_vector_type(8)))  float  v8f;

#define LOG2E 1.4426950408889634f

static __device__ __forceinline__ v8f wmma_bf16(v16bf a, v16bf b, v8f c) {
  // D = A(16x32 bf16) * B(32x16 bf16) + C(16x16 f32)
  return __builtin_amdgcn_wmma_f32_16x16x32_bf16(false, a, false, b, (short)0, c,
                                                 false, false);
}

static __device__ __forceinline__ v16bf zero16() {
  v16bf r;
#pragma unroll
  for (int i = 0; i < 16; ++i) r[i] = (__bf16)0.0f;
  return r;
}

static __device__ __forceinline__ v16bf cat16(v8bf lo, v8bf hi) {
  v16bf r;
#pragma unroll
  for (int i = 0; i < 8; ++i) { r[i] = lo[i]; r[8 + i] = hi[i]; }
  return r;
}

// B fragment (K=32 chunk) from row-major [192][64] bf16 weights in LDS.
// Lane layout: N = lane%16, lanes<16 hold K = 32*chunk+{0..7,16..23},
// lanes>=16 hold K = 32*chunk+{8..15,24..31}.
static __device__ __forceinline__ v16bf loadB(const __bf16* W, int rowBase,
                                              int chunk, int ln, bool hiHalf) {
  const __bf16* p = W + (rowBase + ln) * HID + chunk * 32 + (hiHalf ? 8 : 0);
  return cat16(*(const v8bf*)p, *(const v8bf*)(p + 16));
}

// Wih B fragment: only K rows 0..3 nonzero (D=4, zero-padded to K=32).
// volatile: keep the (cheap) 8-byte load inside the time loop instead of
// letting LICM hoist 12 zero-padded 8-VGPR fragments into live registers.
static __device__ __forceinline__ v16bf loadBx(const __bf16* Wih, int rowBase,
                                               int ln, bool hiHalf) {
  v16bf r = zero16();
  if (!hiHalf) {
    v4bf w = *(const volatile v4bf*)(Wih + (rowBase + ln) * DIM);
#pragma unroll
    for (int i = 0; i < 4; ++i) r[i] = w[i];
  }
  return r;
}

// Fast paths: single v_exp_f32 / v_rcp_f32.
static __device__ __forceinline__ float sigmoidf(float x) {
  return __builtin_amdgcn_rcpf(1.0f + __builtin_amdgcn_exp2f(-LOG2E * x));
}
static __device__ __forceinline__ float tanh_fast(float x) {
  float e = __builtin_amdgcn_exp2f((2.0f * LOG2E) * x);
  return fmaf(-2.0f, __builtin_amdgcn_rcpf(e + 1.0f), 1.0f);
}

// One GRU cell step for a 16-row batch tile owned by this wave.
static __device__ __forceinline__ void gru_step(
    const __bf16* sWhh, const __bf16* sWih,
    const float* sBrz, const float* sBin, const float* sBhn,
    v16bf xa, __bf16* hRow, float (&hC)[4][8], int ln, bool hiHalf) {
  const int klo = hiHalf ? 8 : 0;
  const int khi = hiHalf ? 24 : 16;
  const __bf16* hp = hRow + ln * HID;
  v16bf hA0 = cat16(*(const v8bf*)(hp + klo),      *(const v8bf*)(hp + khi));
  v16bf hA1 = cat16(*(const v8bf*)(hp + 32 + klo), *(const v8bf*)(hp + 32 + khi));

  const v8f zc = {};  // inline-0 C operand

#pragma unroll
  for (int c = 0; c < 4; ++c) {           // hidden-column tile: cols 16c..16c+15
    v8f accR = wmma_bf16(xa,  loadBx(sWih, c * 16, ln, hiHalf), zc);
    accR = wmma_bf16(hA0, loadB(sWhh, c * 16, 0, ln, hiHalf), accR);
    accR = wmma_bf16(hA1, loadB(sWhh, c * 16, 1, ln, hiHalf), accR);

    v8f accZ = wmma_bf16(xa,  loadBx(sWih, 64 + c * 16, ln, hiHalf), zc);
    accZ = wmma_bf16(hA0, loadB(sWhh, 64 + c * 16, 0, ln, hiHalf), accZ);
    accZ = wmma_bf16(hA1, loadB(sWhh, 64 + c * 16, 1, ln, hiHalf), accZ);

    v8f accI = wmma_bf16(xa,  loadBx(sWih, 128 + c * 16, ln, hiHalf), zc);
    v8f accH = wmma_bf16(hA0, loadB(sWhh, 128 + c * 16, 0, ln, hiHalf), zc);
    accH = wmma_bf16(hA1, loadB(sWhh, 128 + c * 16, 1, ln, hiHalf), accH);

    const float bR = sBrz[c * 16 + ln];
    const float bZ = sBrz[64 + c * 16 + ln];
    const float bI = sBin[c * 16 + ln];
    const float bH = sBhn[c * 16 + ln];

#pragma unroll
    for (int v = 0; v < 8; ++v) {
      float r = sigmoidf(accR[v] + bR);
      float z = sigmoidf(accZ[v] + bZ);
      float n = tanh_fast(fmaf(r, accH[v] + bH, accI[v] + bI));
      float h = fmaf(z, hC[c][v] - n, n);   // (1-z)*n + z*h
      hC[c][v] = h;
      hRow[(v + (hiHalf ? 8 : 0)) * HID + c * 16 + ln] = (__bf16)h;
    }
    // Contain scheduler interleaving: one tile's working set live at a time.
    __builtin_amdgcn_sched_barrier(0);
  }
}

__global__ __launch_bounds__(256) void gru_encdec_kernel(
    const float* __restrict__ in,
    const float* __restrict__ eWih, const float* __restrict__ eWhh,
    const float* __restrict__ eBih, const float* __restrict__ eBhh,
    const float* __restrict__ dWih, const float* __restrict__ dWhh,
    const float* __restrict__ dBih, const float* __restrict__ dBhh,
    const float* __restrict__ linW, const float* __restrict__ linB,
    float* __restrict__ out) {
  __shared__ __bf16 sWhhE[192 * HID];
  __shared__ __bf16 sWhhD[192 * HID];
  __shared__ __bf16 sWihE[192 * DIM];
  __shared__ __bf16 sWihD[192 * DIM];
  __shared__ float  sBrzE[128], sBinE[64], sBhnE[64];
  __shared__ float  sBrzD[128], sBinD[64], sBhnD[64];
  __shared__ __bf16 sLinW[DIM * HID];   // bf16 for the WMMA head
  __shared__ float  sLinB[DIM];
  __shared__ __bf16 sH[8][16 * HID];    // per-wave h staging (row-major)
  __shared__ float  sPx[8][16 * DIM];   // per-wave decoder feedback

  const int tid = threadIdx.x;
  for (int i = tid; i < 192 * HID; i += 256) {
    sWhhE[i] = (__bf16)eWhh[i];
    sWhhD[i] = (__bf16)dWhh[i];
  }
  for (int i = tid; i < 192 * DIM; i += 256) {
    sWihE[i] = (__bf16)eWih[i];
    sWihD[i] = (__bf16)dWih[i];
  }
  for (int i = tid; i < 128; i += 256) {
    sBrzE[i] = eBih[i] + eBhh[i];
    sBrzD[i] = dBih[i] + dBhh[i];
  }
  for (int i = tid; i < 64; i += 256) {
    sBinE[i] = eBih[128 + i];
    sBhnE[i] = eBhh[128 + i];
    sBinD[i] = dBih[128 + i];
    sBhnD[i] = dBhh[128 + i];
  }
  for (int i = tid; i < DIM * HID; i += 256) sLinW[i] = (__bf16)linW[i];
  if (tid < DIM) sLinB[tid] = linB[tid];
  for (int i = tid; i < 8 * 16 * HID; i += 256) (&sH[0][0])[i] = (__bf16)0.0f;
  __syncthreads();

  const int  wave   = tid >> 5;
  const int  lane   = tid & 31;
  const int  ln     = lane & 15;
  const bool hiHalf = lane >= 16;
  const int  klo    = hiHalf ? 8 : 0;
  const int  khi    = hiHalf ? 24 : 16;
  __bf16* hRow  = &sH[wave][0];
  float*  pxRow = &sPx[wave][0];
  const size_t rowBase0 = (size_t)blockIdx.x * 128 + (size_t)wave * 16;
  const size_t rowG     = rowBase0 + ln;

  float hC[4][8];
#pragma unroll
  for (int c = 0; c < 4; ++c)
#pragma unroll
    for (int v = 0; v < 8; ++v) hC[c][v] = 0.0f;

  const float* inRow = in + rowG * (size_t)(T_SEQ * DIM);
  float4 prev  = make_float4(0.f, 0.f, 0.f, 0.f);
  float4 lastX = make_float4(0.f, 0.f, 0.f, 0.f);
  if (!hiHalf) prev = *(const float4*)inRow;

  v16bf xa = zero16();  // loop-carried; only elems 0..3 rewritten (lanes<16)

  // ---------------- encoder: 50 steps over diffs ----------------
#pragma unroll 1
  for (int t = 0; t < T_SEQ - 1; ++t) {
    if (!hiHalf) {
      float4 cur = *(const float4*)(inRow + (size_t)(t + 1) * DIM);
      lastX = make_float4(cur.x - prev.x, cur.y - prev.y,
                          cur.z - prev.z, cur.w - prev.w);
      prev = cur;
      xa[0] = (__bf16)lastX.x; xa[1] = (__bf16)lastX.y;
      xa[2] = (__bf16)lastX.z; xa[3] = (__bf16)lastX.w;
    }
    gru_step(sWhhE, sWihE, sBrzE, sBinE, sBhnE, xa, hRow, hC, ln, hiHalf);
  }

  // decoder feedback init: px = last diff
  if (!hiHalf) *(float4*)(pxRow + ln * DIM) = lastX;

  // Head state kept in C-layout by the 8 lanes with ln<4:
  // lane holds output dim d=ln for rows M = v + 8*hiHalf.
  const bool headLane = (ln < DIM);
  float pxC[8], offC[8];
  const float lb = sLinB[ln & (DIM - 1)];
#pragma unroll
  for (int v = 0; v < 8; ++v) { pxC[v] = 0.0f; offC[v] = 0.0f; }
  if (headLane) {
#pragma unroll
    for (int v = 0; v < 8; ++v) {
      int rowM = v + (hiHalf ? 8 : 0);
      pxC[v]  = pxRow[rowM * DIM + ln];
      offC[v] = in[(rowBase0 + rowM) * (size_t)(T_SEQ * DIM) +
                   (size_t)(T_SEQ - 1) * DIM + ln];
    }
  }

  // linW B fragments (single N-tile, rows 0..3 valid), hoisted: 16 VGPRs
  v16bf lb0 = zero16(), lb1 = zero16();
  if (headLane) {
    const __bf16* p0 = sLinW + ln * HID + klo;
    lb0 = cat16(*(const v8bf*)p0, *(const v8bf*)(sLinW + ln * HID + khi));
    const __bf16* p1 = sLinW + ln * HID + 32 + klo;
    lb1 = cat16(*(const v8bf*)p1, *(const v8bf*)(sLinW + ln * HID + 32 + khi));
  }
  const v8f zc = {};

  // ---------------- decoder: 20 autoregressive steps ----------------
#pragma unroll 1
  for (int s = 0; s < NOUT; ++s) {
    if (!hiHalf) {
      float4 px = *(const float4*)(pxRow + ln * DIM);
      xa[0] = (__bf16)px.x; xa[1] = (__bf16)px.y;
      xa[2] = (__bf16)px.z; xa[3] = (__bf16)px.w;
    }
    gru_step(sWhhD, sWihD, sBrzD, sBinD, sBhnD, xa, hRow, hC, ln, hiHalf);

    // linear head via WMMA: x = h @ linW^T + lin_b + px
    const __bf16* hp = hRow + ln * HID;
    v16bf hN0 = cat16(*(const v8bf*)(hp + klo),      *(const v8bf*)(hp + khi));
    v16bf hN1 = cat16(*(const v8bf*)(hp + 32 + klo), *(const v8bf*)(hp + 32 + khi));
    v8f ho = wmma_bf16(hN0, lb0, zc);
    ho = wmma_bf16(hN1, lb1, ho);

    if (headLane) {
#pragma unroll
      for (int v = 0; v < 8; ++v) {
        int rowM = v + (hiHalf ? 8 : 0);
        float x = ho[v] + lb + pxC[v];
        pxC[v] = x;
        pxRow[rowM * DIM + ln] = x;  // feedback for next step's xa
        out[((rowBase0 + rowM) * NOUT + (size_t)s) * DIM + ln] = x + offC[v];
      }
    }
  }
}

extern "C" void kernel_launch(void* const* d_in, const int* in_sizes, int n_in,
                              void* d_out, int out_size, void* d_ws, size_t ws_size,
                              hipStream_t stream) {
  (void)in_sizes; (void)n_in; (void)out_size; (void)d_ws; (void)ws_size;
  const float* in   = (const float*)d_in[0];
  const float* eWih = (const float*)d_in[1];
  const float* eWhh = (const float*)d_in[2];
  const float* eBih = (const float*)d_in[3];
  const float* eBhh = (const float*)d_in[4];
  const float* dWih = (const float*)d_in[5];
  const float* dWhh = (const float*)d_in[6];
  const float* dBih = (const float*)d_in[7];
  const float* dBhh = (const float*)d_in[8];
  const float* linW = (const float*)d_in[9];
  const float* linB = (const float*)d_in[10];

  dim3 grid(B_TOTAL / 128);
  dim3 block(256);
  gru_encdec_kernel<<<grid, block, 0, stream>>>(in, eWih, eWhh, eBih, eBhh,
                                                dWih, dWhh, dBih, dBhh,
                                                linW, linB, (float*)d_out);
}
